// AttentionCommModule_28870770164061
// MI455X (gfx1250) — matrix operand order repaired and checked by reading, not compiled, not run on hardware
//
#include <hip/hip_runtime.h>
#include <stdint.h>

// ---------------- problem constants ----------------
#define IN_DIM   256
#define CD       128           // comm dim
#define NH       4             // heads
#define DH       32            // head dim
#define NAG      8             // agents (1 self + 7 msgs)
#define TM       16            // batch rows per tile
#define KC       32            // K per bf16 WMMA
#define NCHUNK   (IN_DIM / KC) // 8
#define BROWS    32768
#define SCALE    0.17677669529663687f   // 1/sqrt(32)

typedef __attribute__((ext_vector_type(16))) __bf16 v16bf;
typedef __attribute__((ext_vector_type(8)))  __bf16 v8bf;
typedef __attribute__((ext_vector_type(4)))  __bf16 v4bf;
typedef __attribute__((ext_vector_type(8)))  float  v8f;
typedef __attribute__((ext_vector_type(4)))  unsigned int u32x4;
typedef __attribute__((ext_vector_type(8)))  int    i32x8;
typedef __attribute__((ext_vector_type(4)))  int    i32x4;

// ---------------- LDS layout (bytes) ----------------
// X:   128 rows x 256 bf16  (all_msgs tile, agent-major rows r = n*16 + m)
// K,V: per-wave head tiles [4][128 x 32] bf16
// Q:   per-wave [4][16 x 32] f32
// WGT: weighted 16x128 bf16, overlays X   (X dead after projections)
// WO:  staged Wo 128x128 bf16, overlays K (K dead after attention; TDM'd late,
//      wave w's Wo rows 32w..32w+31 land exactly in wave w's K slice)
// Total 136 KB -> 2 workgroups per 320 KB WGP (8 waves resident).
#define OFF_X    0
#define OFF_K    65536
#define OFF_V    (OFF_K + 32768)
#define OFF_Q    (OFF_V + 32768)
#define SMEM_BYTES (OFF_Q + 8192)     // 139264 B = 136 KB
#define OFF_WGT  OFF_X
#define OFF_WO   OFF_K

// ---------------- gfx1250 TDM availability ----------------
#if defined(__gfx1250__) && __has_builtin(__builtin_amdgcn_tensor_load_to_lds) && \
    __has_builtin(__builtin_amdgcn_s_wait_tensorcnt)
#define USE_TDM 1
#else
#define USE_TDM 0
#endif

#if USE_TDM
// 1-D contiguous copy of nelems (2-byte elements) global -> LDS via Tensor Data Mover.
__device__ __forceinline__ void tdm_load_1d_b16(unsigned int lds_byte_addr,
                                                const void* gptr, int nelems) {
  uint64_t ga = (uint64_t)(uintptr_t)gptr;
  u32x4 g0;
  g0[0] = 1u;                                   // count=1 valid descriptor, user mode
  g0[1] = lds_byte_addr;                        // lds_addr (bytes)
  g0[2] = (unsigned int)ga;                     // global_addr[31:0]
  g0[3] = ((unsigned int)(ga >> 32) & 0x01FFFFFFu) | (2u << 30);  // addr[56:32] | type=2
  i32x8 g1;
  g1[0] = (1 << 16);                            // wg_mask=0, data_size=1 (2 bytes)
  g1[1] = (nelems & 0xFFFF) << 16;              // tensor_dim0 lo16
  g1[2] = ((nelems >> 16) & 0xFFFF) | (1 << 16);// tensor_dim0 hi16 | tensor_dim1=1
  g1[3] = (nelems & 0xFFFF) << 16;              // tile_dim0
  g1[4] = 1;                                    // tile_dim1=1, tile_dim2=0
  g1[5] = nelems;                               // tensor_dim0_stride lo32
  g1[6] = 0;
  g1[7] = 0;
  i32x4 gz = {0, 0, 0, 0};
#if defined(__clang_major__) && (__clang_major__ >= 23)
  i32x8 gz8 = {0, 0, 0, 0, 0, 0, 0, 0};
  __builtin_amdgcn_tensor_load_to_lds(g0, g1, gz, gz, gz8, 0);
#else
  __builtin_amdgcn_tensor_load_to_lds(g0, g1, gz, gz, 0);
#endif
}
#endif

// ---------------- WMMA fragment helpers ----------------
// Load a 16-element bf16 fragment as two 16B chunks at +0 and +hiOff elements.
// A-frag (16x32):  p = row*256 + chunk*32 + (lane>=16? 8:0), hiOff = 16
// B-frag (32x16):  p = col*K   + chunk*32 + (lane>=16? 8:0), hiOff = 16
__device__ __forceinline__ v16bf ld_frag(const __bf16* p, int hiOff) {
  v8bf lo = *(const v8bf*)p;
  v8bf hi = *(const v8bf*)(p + hiOff);
  return __builtin_shufflevector(lo, hi, 0, 1, 2, 3, 4, 5, 6, 7,
                                         8, 9, 10, 11, 12, 13, 14, 15);
}

__device__ __forceinline__ v8f wmma_bf16(v16bf a, v16bf b, v8f c) {
  return __builtin_amdgcn_wmma_f32_16x16x32_bf16(false, a, false, b,
                                                 (short)0, c, false, false);
}

// ---------------- prep: weights f32 -> bf16 in workspace ----------------
// ws layout (elements): [Wq 32768][Wk 32768][Wv 32768][Wo 16384], original [out][in]
__global__ void convert_weights_kernel(const float* __restrict__ wq,
                                       const float* __restrict__ wk,
                                       const float* __restrict__ wv,
                                       const float* __restrict__ wo,
                                       __bf16* __restrict__ dst) {
  int i = blockIdx.x * blockDim.x + threadIdx.x;
  if (i < 32768)       dst[i] = (__bf16)wq[i];
  else if (i < 65536)  dst[i] = (__bf16)wk[i - 32768];
  else if (i < 98304)  dst[i] = (__bf16)wv[i - 65536];
  else if (i < 114688) dst[i] = (__bf16)wo[i - 98304];
}

// ---------------- fused attention-comm kernel ----------------
__global__ __launch_bounds__(128, 2) void attn_comm_kernel(
    const float* __restrict__ obs, const float* __restrict__ msgs,
    const float* __restrict__ bq, const float* __restrict__ bk,
    const float* __restrict__ bv, const float* __restrict__ bo,
    const __bf16* __restrict__ wqb, const __bf16* __restrict__ wkb,
    const __bf16* __restrict__ wvb, const __bf16* __restrict__ wob,
    float* __restrict__ out) {
  __shared__ __align__(16) unsigned char smem[SMEM_BYTES];

  const int tid  = threadIdx.x;
  const int lane = tid & 31;
  const int wid  = tid >> 5;          // wave id == head id (4 waves, wave32)
  const int l15  = lane & 15;
  const int hl   = lane >> 4;         // lane half selects K sub-range of fragment
  const int b0   = blockIdx.x * TM;   // first batch row of this tile

  __bf16* X   = (__bf16*)(smem + OFF_X);
  __bf16* Ksh = (__bf16*)(smem + OFF_K) + wid * (NAG * TM * DH);
  __bf16* Vsh = (__bf16*)(smem + OFF_V) + wid * (NAG * TM * DH);
  float*  Qsh = (float*) (smem + OFF_Q) + wid * (TM * DH);
  __bf16* WOs = (__bf16*)(smem + OFF_WO);
  __bf16* Wgt = (__bf16*)(smem + OFF_WGT);

  // ---- stage all_msgs tile: 128 rows x 256 f32 -> bf16 (read each input byte once)
  for (int i = tid; i < (NAG * TM) * (IN_DIM / 4); i += 128) {
    int r  = i >> 6;                  // row = n*16 + m
    int c4 = i & 63;
    int m = r & 15, n = r >> 4;
    const float* src = (n == 0)
        ? (obs + (size_t)(b0 + m) * IN_DIM)
        : (msgs + ((size_t)(b0 + m) * (NAG - 1) + (n - 1)) * IN_DIM);
    float4 v = ((const float4*)src)[c4];
    v4bf o;
    o.x = (__bf16)v.x; o.y = (__bf16)v.y; o.z = (__bf16)v.z; o.w = (__bf16)v.w;
    *(v4bf*)(X + r * IN_DIM + c4 * 4) = o;
  }
  __syncthreads();

  // ---- K/V projections for this wave's head: 4 combos = {K,V} x {lo,hi 16 cols}
  // B-fragment held across the agent loop (weights re-used in registers, L2-light).
  // A-fragments for all 8 agents preloaded per chunk so the 16 ds_load_b128 issue
  // as a clause and drain across the 8-WMMA burst (explicit reuse distance keeps
  // pipelining alive under the 2-blocks/SIMD register budget).
  for (int c = 0; c < 4; ++c) {
    const int th = c & 1;
    const __bf16* W    = (c < 2) ? wkb : wvb;
    const float*  bias = (c < 2) ? bk  : bv;
    __bf16*       dst  = (c < 2) ? Ksh : Vsh;
    const int col = wid * DH + th * 16 + l15;   // output column (0..127)
    const float bc = bias[col];
    v8f C[NAG];
#pragma unroll
    for (int n = 0; n < NAG; ++n) {
#pragma unroll
      for (int j = 0; j < 8; ++j) C[n][j] = bc;
    }
    for (int ch = 0; ch < NCHUNK; ++ch) {
      v16bf B = ld_frag(W + (size_t)col * IN_DIM + ch * KC + hl * 8, 16);
      v16bf A[NAG];
#pragma unroll
      for (int n = 0; n < NAG; ++n)
        A[n] = ld_frag(X + (n * TM + l15) * IN_DIM + ch * KC + hl * 8, 16);
#pragma unroll
      for (int n = 0; n < NAG; ++n)
        C[n] = wmma_bf16(A[n], B, C[n]);
    }
#pragma unroll
    for (int n = 0; n < NAG; ++n) {
#pragma unroll
      for (int j = 0; j < 8; ++j)
        dst[(n * TM + j + hl * 8) * DH + th * 16 + l15] = (__bf16)C[n][j];
    }
  }

  // ---- Q projection (agent_obs = X rows 0..15), this wave's head columns
#pragma unroll
  for (int th = 0; th < 2; ++th) {
    const int col = wid * DH + th * 16 + l15;
    const float bc = bq[col];
    v8f C;
#pragma unroll
    for (int j = 0; j < 8; ++j) C[j] = bc;
    for (int ch = 0; ch < NCHUNK; ++ch) {
      v16bf B = ld_frag(wqb + (size_t)col * IN_DIM + ch * KC + hl * 8, 16);
      v16bf A = ld_frag(X + l15 * IN_DIM + ch * KC + hl * 8, 16);
      C = wmma_bf16(A, B, C);
    }
#pragma unroll
    for (int j = 0; j < 8; ++j)
      Qsh[(j + hl * 8) * DH + th * 16 + l15] = C[j];
  }
  __syncthreads();   // all waves done reading X; weighted buffer may overlay it

  // ---- single-query attention for this head (VALU; ~0.4% of total FLOPs)
  if (lane < TM) {
    const int m = lane;
    float q[DH];
    for (int d = 0; d < DH; ++d) q[d] = Qsh[m * DH + d];
    float s[NAG];
    float mx = -3.0e38f;
    for (int n = 0; n < NAG; ++n) {
      float acc = 0.f;
      for (int d = 0; d < DH; ++d)
        acc += q[d] * (float)Ksh[(n * TM + m) * DH + d];
      s[n] = acc * SCALE;
      mx = fmaxf(mx, s[n]);
    }
    float sum = 0.f;
    for (int n = 0; n < NAG; ++n) { s[n] = __expf(s[n] - mx); sum += s[n]; }
    const float inv = 1.0f / sum;
    float w[DH];
    for (int d = 0; d < DH; ++d) w[d] = 0.f;
    for (int n = 0; n < NAG; ++n) {
      const float p = s[n] * inv;
      for (int d = 0; d < DH; ++d)
        w[d] += p * (float)Vsh[(n * TM + m) * DH + d];
    }
    for (int d = 0; d < DH; ++d)
      Wgt[m * CD + wid * DH + d] = (__bf16)w[d];
  }
  __syncthreads();   // K/V now dead everywhere; Wo staging may overlay K

  // ---- stage Wo (bf16, 128x128) into the dead K region: one TDM descriptor per
  //      wave; wave w copies exactly the 32 rows (cols 32w..32w+31) it consumes.
#if USE_TDM
  {
    unsigned int laddr = (unsigned int)(uintptr_t)(void*)(smem + OFF_WO) + wid * 8192;
    tdm_load_1d_b16(laddr, (const void*)(wob + wid * 4096), 4096);
    __builtin_amdgcn_s_wait_tensorcnt(0);
  }
#else
  for (int i = tid; i < (CD * CD * 2) / 16; i += 128)
    ((uint4*)(smem + OFF_WO))[i] = ((const uint4*)wob)[i];
#endif
  __syncthreads();

  // ---- output projection: weighted(16x128) @ Wo^T + bo, wave does cols 32w..32w+31
#pragma unroll
  for (int th = 0; th < 2; ++th) {
    const int col = wid * DH + th * 16 + l15;
    const float bc = bo[col];
    v8f C;
#pragma unroll
    for (int j = 0; j < 8; ++j) C[j] = bc;
    for (int ch = 0; ch < CD / KC; ++ch) {   // 4 chunks of K=32
      v16bf B = ld_frag(WOs + col * CD + ch * KC + hl * 8, 16);
      v16bf A = ld_frag(Wgt + l15 * CD + ch * KC + hl * 8, 16);
      C = wmma_bf16(A, B, C);
    }
#pragma unroll
    for (int j = 0; j < 8; ++j)
      out[(size_t)(b0 + j + hl * 8) * CD + col] = C[j];
  }
}

// ---------------- host launcher ----------------
extern "C" void kernel_launch(void* const* d_in, const int* in_sizes, int n_in,
                              void* d_out, int out_size, void* d_ws, size_t ws_size,
                              hipStream_t stream) {
  const float* obs = (const float*)d_in[0];
  const float* msgs = (const float*)d_in[1];
  const float* Wq = (const float*)d_in[2];
  const float* bq = (const float*)d_in[3];
  const float* Wk = (const float*)d_in[4];
  const float* bk = (const float*)d_in[5];
  const float* Wv = (const float*)d_in[6];
  const float* bv = (const float*)d_in[7];
  const float* Wo = (const float*)d_in[8];
  const float* bo = (const float*)d_in[9];

  __bf16* wbf = (__bf16*)d_ws;   // 114688 bf16 elems = 224 KB of workspace

  convert_weights_kernel<<<448, 256, 0, stream>>>(Wq, Wk, Wv, Wo, wbf);
  attn_comm_kernel<<<BROWS / TM, 128, 0, stream>>>(
      obs, msgs, bq, bk, bv, bo,
      wbf, wbf + 32768, wbf + 65536, wbf + 98304,
      (float*)d_out);
}
